// MixedScore_MultiHeadAttention_17978733101107
// MI455X (gfx1250) — matrix operand order repaired and checked by reading, not compiled, not run on hardware
//
#include <hip/hip_runtime.h>
#include <hip/hip_bf16.h>

typedef __attribute__((ext_vector_type(16))) _Float16 v16h;
typedef __attribute__((ext_vector_type(8)))  _Float16 v8h;
typedef __attribute__((ext_vector_type(8)))  float    v8f;

#define B_   4
#define H_   16
#define T_   512
#define M_   512
#define D_   64
#define HID_ 16
#define TT_  32           // T rows per block (2 row-tiles of 16)
#define NWAVE 8
#define NTHR 256
#define MPAD (M_ + 4)     // f32 row pad: keeps 16B alignment, spreads banks
#define VPAD (M_ + 8)     // f16 row pad

__global__ __launch_bounds__(NTHR) void msmha_kernel(
    const float* __restrict__ q,  const float* __restrict__ k,
    const float* __restrict__ v,  const float* __restrict__ Dtm,
    const float* __restrict__ w1, const float* __restrict__ b1,
    const float* __restrict__ w2, const float* __restrict__ b2,
    float* __restrict__ out)
{
    const int t0   = blockIdx.x * TT_;
    const int h    = blockIdx.y;
    const int b    = blockIdx.z;
    const int tid  = threadIdx.x;
    const int wave = tid >> 5;
    const int lane = tid & 31;
    const int lh   = lane >> 4;     // lane half (0/1)
    const int ln   = lane & 15;
    const int rt   = wave >> 2;     // row-tile 0/1 (16 rows each)
    const int wq   = wave & 3;      // quarter within row-tile

    __shared__ float    s_p[TT_][MPAD];     // scores -> mixed -> exp(p)
    __shared__ _Float16 s_vt[D_][VPAD];     // V transposed, f16
    __shared__ float    s_red[TT_][8];
    __shared__ float    s_red2[TT_][8];
    __shared__ float    s_inv[TT_];

    const size_t bh   = (size_t)b * H_ + h;
    const float* qrow = q + (bh * T_ + t0 + rt * 16) * (size_t)D_;
    const float* kb   = k + bh * (size_t)M_ * D_;
    const float* vb   = v + bh * (size_t)M_ * D_;

    // ---- stage V into LDS, transposed (d-major) and converted to f16 ----
    {
        const float4* v4 = (const float4*)vb;
        #pragma unroll
        for (int it = 0; it < (M_ * D_ / 4) / NTHR; ++it) {   // 32 iters
            const int idx = it * NTHR + tid;
            const int m   = idx >> 4;            // 16 float4 per V row
            const int d4  = (idx & 15) * 4;
            const float4 val = v4[idx];
            s_vt[d4 + 0][m] = (_Float16)val.x;
            s_vt[d4 + 1][m] = (_Float16)val.y;
            s_vt[d4 + 2][m] = (_Float16)val.z;
            s_vt[d4 + 3][m] = (_Float16)val.w;
        }
    }

    // ---- build Q A-fragments (2 chunks of d) for this wave's row-tile ----
    // A 16x32 f16 layout: lane row = ln, K runs [8*lh,+8) and [16+8*lh,+8)
    v16h aq[2];
    #pragma unroll
    for (int c = 0; c < 2; ++c) {
        const float* p0 = qrow + (size_t)ln * D_ + c * 32 + 8 * lh;
        const float* p1 = p0 + 16;
        #pragma unroll
        for (int e = 0; e < 8; ++e) {
            aq[c][e]     = (_Float16)p0[e];
            aq[c][8 + e] = (_Float16)p1[e];
        }
    }

    // ---- phase 1: S = Q K^T ; wave (rt,wq) covers m-tiles wq,wq+4,... ----
    // uniform (scalar) trip count: EXEC stays all-1s through the WMMAs
    #pragma unroll 1
    for (int it = 0; it < (M_ / 16) / 4; ++it) {
        const int m0 = (wq + it * 4) * 16;
        v8f acc = {};
        #pragma unroll
        for (int c = 0; c < 2; ++c) {
            // B 32x16 f16 layout: lane col = ln, K runs [8*lh,+8) and [16+8*lh,+8)
            const float* kp = kb + (size_t)(m0 + ln) * D_ + c * 32 + 8 * lh;
            v16h bk;
            #pragma unroll
            for (int e = 0; e < 16; ++e) bk[e] = (_Float16)kp[e < 8 ? e : e + 8];
            acc = __builtin_amdgcn_wmma_f32_16x16x32_f16(false, aq[c], false, bk,
                                                         (short)0, acc, false, false);
        }
        #pragma unroll
        for (int g = 0; g < 8; ++g)
            s_p[rt * 16 + g + 8 * lh][m0 + ln] = acc[g];
    }
    __syncthreads();

    // ---- phase 2: per-element 2->16->1 ReLU MLP + softmax statistics ----
    float w1d[HID_], w1x[HID_], bb1[HID_], ww2[HID_];
    #pragma unroll
    for (int c = 0; c < HID_; ++c) {
        w1d[c] = w1[(h * 2 + 0) * HID_ + c] * 0.125f;   // fold 1/sqrt(d)
        w1x[c] = w1[(h * 2 + 1) * HID_ + c];
        bb1[c] = b1[h * HID_ + c];
        ww2[c] = w2[h * HID_ + c];
    }
    const float bb2 = b2[h];

    const int r  = tid >> 3;         // row 0..31
    const int sg = tid & 7;          // 64-wide segment 0..7
    const int ms = sg * 64;
    const float* drow = Dtm + ((size_t)b * T_ + (t0 + r)) * M_ + ms;

    float lmax = -3.4e38f;
    for (int i = 0; i < 64; i += 4) {
        const float4 s4 = *(const float4*)&s_p[r][ms + i];
        const float4 d4 = *(const float4*)(drow + i);
        float sv[4] = {s4.x, s4.y, s4.z, s4.w};
        float dv[4] = {d4.x, d4.y, d4.z, d4.w};
        float mx[4];
        #pragma unroll
        for (int j = 0; j < 4; ++j) {
            float acc = bb2;
            #pragma unroll
            for (int c = 0; c < HID_; ++c) {
                float t = fmaf(sv[j], w1d[c], fmaf(dv[j], w1x[c], bb1[c]));
                acc = fmaf(fmaxf(t, 0.f), ww2[c], acc);
            }
            mx[j] = acc;
            lmax  = fmaxf(lmax, acc);
        }
        *(float4*)&s_p[r][ms + i] = make_float4(mx[0], mx[1], mx[2], mx[3]);
    }
    s_red[r][sg] = lmax;
    __syncthreads();
    float rmax = s_red[r][0];
    #pragma unroll
    for (int j = 1; j < 8; ++j) rmax = fmaxf(rmax, s_red[r][j]);

    float lsum = 0.f;
    for (int i = 0; i < 64; i += 4) {
        const float4 s4 = *(const float4*)&s_p[r][ms + i];
        float p0 = __expf(s4.x - rmax);
        float p1 = __expf(s4.y - rmax);
        float p2 = __expf(s4.z - rmax);
        float p3 = __expf(s4.w - rmax);
        lsum += (p0 + p1) + (p2 + p3);
        *(float4*)&s_p[r][ms + i] = make_float4(p0, p1, p2, p3);
    }
    s_red2[r][sg] = lsum;
    __syncthreads();
    if (sg == 0) {
        float rs = 0.f;
        #pragma unroll
        for (int j = 0; j < 8; ++j) rs += s_red2[r][j];
        s_inv[r] = 1.0f / rs;               // fold softmax norm into final store
    }
    __syncthreads();

    // ---- phase 3: O = P V ; wave (rt,dt=wq) owns a 16x16 output tile ----
    const int dt = wq;
    v8f oacc = {};
    #pragma unroll 1
    for (int mc = 0; mc < M_ / 32; ++mc) {
        const int m0 = mc * 32;
        v16h ap;
        {
            // contiguous aligned runs -> ds_load_b128 x4
            const float4* pr = (const float4*)&s_p[rt * 16 + ln][m0 + 8 * lh];
            const float4 f0 = pr[0], f1 = pr[1], f2 = pr[4], f3 = pr[5];
            ap[0]  = (_Float16)f0.x; ap[1]  = (_Float16)f0.y;
            ap[2]  = (_Float16)f0.z; ap[3]  = (_Float16)f0.w;
            ap[4]  = (_Float16)f1.x; ap[5]  = (_Float16)f1.y;
            ap[6]  = (_Float16)f1.z; ap[7]  = (_Float16)f1.w;
            ap[8]  = (_Float16)f2.x; ap[9]  = (_Float16)f2.y;
            ap[10] = (_Float16)f2.z; ap[11] = (_Float16)f2.w;
            ap[12] = (_Float16)f3.x; ap[13] = (_Float16)f3.y;
            ap[14] = (_Float16)f3.z; ap[15] = (_Float16)f3.w;
        }
        v16h bv;
        {
            // B 32x16: lane col = ln -> row d = dt*16+ln of s_vt; 16B aligned runs at 8*lh and 16+8*lh
            const v8h* vp = (const v8h*)&s_vt[dt * 16 + ln][m0 + 8 * lh];
            const v8h lo = vp[0], hi = vp[2];   // ds_load_b128 x2
            bv = __builtin_shufflevector(lo, hi, 0, 1, 2, 3, 4, 5, 6, 7,
                                         8, 9, 10, 11, 12, 13, 14, 15);
        }
        oacc = __builtin_amdgcn_wmma_f32_16x16x32_f16(false, ap, false, bv,
                                                      (short)0, oacc, false, false);
    }

    // out shape (B, T, H*D): row stride H*D
    float* ob = out + ((size_t)b * T_ + t0 + rt * 16) * (H_ * D_) + h * D_ + dt * 16 + ln;
    #pragma unroll
    for (int g = 0; g < 8; ++g) {
        const int tl = g + 8 * lh;
        ob[(size_t)tl * (H_ * D_)] = oacc[g] * s_inv[rt * 16 + tl];
    }
}

extern "C" void kernel_launch(void* const* d_in, const int* in_sizes, int n_in,
                              void* d_out, int out_size, void* d_ws, size_t ws_size,
                              hipStream_t stream) {
    (void)in_sizes; (void)n_in; (void)out_size; (void)d_ws; (void)ws_size;
    const float* q   = (const float*)d_in[0];
    const float* k   = (const float*)d_in[1];
    const float* v   = (const float*)d_in[2];
    const float* Dtm = (const float*)d_in[3];
    const float* w1  = (const float*)d_in[4];
    const float* b1  = (const float*)d_in[5];
    const float* w2  = (const float*)d_in[6];
    const float* b2  = (const float*)d_in[7];
    float* out = (float*)d_out;

    dim3 grid(T_ / TT_, H_, B_);   // (16, 16, 4)
    dim3 block(NTHR);
    msmha_kernel<<<grid, block, 0, stream>>>(q, k, v, Dtm, w1, b1, w2, b2, out);
}
